// RPN_77343771066937
// MI455X (gfx1250) — matrix operand order repaired and checked by reading, not compiled, not run on hardware
//
#include <hip/hip_runtime.h>
#include <math.h>

typedef unsigned short u16;
typedef unsigned int   u32;
typedef unsigned long long u64;

typedef __attribute__((ext_vector_type(16))) __bf16 v16bf;
typedef __attribute__((ext_vector_type(8)))  float  v8f;
typedef int v4i __attribute__((vector_size(16)));   // GCC-vector int4: matches builtin proto V4i

#define LSTR 40   // LDS row stride in u16 (80B: keeps 16B alignment, staggers banks)
#define NB   94   // ceil(6000/64)

// ---- CDNA5 async global->LDS path (probe via __has_builtin; safe fallback) ----
#if defined(__HIP_DEVICE_COMPILE__)
#  if __has_builtin(__builtin_amdgcn_global_load_async_to_lds_b128)
#    define HAVE_ASYNC_LDS 1
     // AS casts routed through integers (clang forbids direct C-style AS casts in HIP).
     // Low 32 bits of a generic LDS pointer == LDS offset (AMDGCN addrspacecast semantics).
#    define GLB_AS1(p) ((__attribute__((address_space(1))) v4i*)(unsigned long long)(p))
#    define LDS_AS3(p) ((__attribute__((address_space(3))) v4i*)(unsigned)(unsigned long long)(p))
#    define ASYNC_CP16(ldsp, gp)                                             \
       __builtin_amdgcn_global_load_async_to_lds_b128(GLB_AS1(gp), LDS_AS3(ldsp), 0, 0)
#  endif
#  if __has_builtin(__builtin_amdgcn_s_wait_asynccnt)
#    define WAIT_ASYNC() __builtin_amdgcn_s_wait_asynccnt(0)
#  else
#    define WAIT_ASYNC() asm volatile("s_wait_asynccnt 0x0" ::: "memory")
#  endif
#else
#  define WAIT_ASYNC() ((void)0)
#endif

__device__ __forceinline__ u16 f2bf(float f) {
  u32 u = __float_as_uint(f);
  u += 0x7FFFu + ((u >> 16) & 1u);       // round-to-nearest-even
  return (u16)(u >> 16);
}

// ---------------- packing kernels ----------------
// conv_w (O=1024,I=1024,3,3) f32 -> Wt[p][o][ci] bf16
__global__ void k_prep_w(const float* __restrict__ w, u16* __restrict__ Wt) {
  int gid = blockIdx.x * 256 + threadIdx.x;          // 9*1024*1024
  int ci = gid & 1023;
  int o  = (gid >> 10) & 1023;
  int p  = gid >> 20;
  if (p < 9) Wt[gid] = f2bf(w[((size_t)(o * 1024 + ci)) * 9 + p]);
}

// x (C=1024,64,64) f32 -> Xt[s][ci] bf16
__global__ void k_prep_x(const float* __restrict__ x, u16* __restrict__ Xt) {
  int gid = blockIdx.x * 256 + threadIdx.x;          // 4096*1024
  int ci = gid & 1023;
  int s  = gid >> 10;
  Xt[gid] = f2bf(x[(size_t)ci * 4096 + s]);
}

// score_w (15,1024) + loc_w (60,1024) -> Wsl[80][1024] bf16
__global__ void k_prep_wsl(const float* __restrict__ sw, const float* __restrict__ lw,
                           u16* __restrict__ Wsl) {
  int gid = blockIdx.x * 256 + threadIdx.x;          // 80*1024
  int ci = gid & 1023;
  int n  = gid >> 10;
  float v = 0.f;
  if (n < 15)      v = sw[n * 1024 + ci];
  else if (n < 75) v = lw[(n - 15) * 1024 + ci];
  Wsl[gid] = f2bf(v);
}

// ---------------- 3x3 conv as implicit GEMM (bf16 WMMA, async-LDS double buffer) --------
__global__ __launch_bounds__(256) void k_conv3x3(
    const u16* __restrict__ Xt, const u16* __restrict__ Wt,
    const float* __restrict__ bias, u16* __restrict__ Ht) {
  __shared__ u16 As[2][128 * LSTR];
  __shared__ u16 Bs[2][128 * LSTR];
  const int tid  = threadIdx.x;
  const int lane = tid & 31;
  const int wave = tid >> 5;
  const int n0 = blockIdx.x * 128;                   // out-channel tile
  const int m0 = blockIdx.y * 128;                   // spatial tile
  const int wm = (wave & 1) * 64;
  const int wn = (wave >> 1) * 32;

  v8f zero = {0.f, 0.f, 0.f, 0.f, 0.f, 0.f, 0.f, 0.f};
  v8f acc[4][2];
#pragma unroll
  for (int i = 0; i < 4; ++i)
#pragma unroll
    for (int j = 0; j < 2; ++j) acc[i][j] = zero;

  const int lrow  = tid >> 1;                        // 0..127 staging row
  const int lhalf = (tid & 1) * 16;                  // K sub-offset 0/16
  const int sy = (m0 + lrow) >> 6;
  const int sx = (m0 + lrow) & 63;

  const int am = lane & 15;
  const int ak = (lane >> 4) * 8;                    // A frag K base 2*ak = 16h (contiguous k = 16h + s)
  const int bk = (lane >> 4) * 16;                   // B frag K base 16h (contiguous k = 16h + s, same as A)

  union F { uint4 u[2]; v16bf v; };

  // stage step s (s = p*32 + kchunk) into LDS buffer `buf`
  auto stage = [&](int s, int buf) {
    const int p  = s >> 5;
    const int kc = (s & 31) << 5;
    const int yy = sy + p / 3 - 1;
    const int xx = sx + p % 3 - 1;
    u16* ad = &As[buf][lrow * LSTR + lhalf];
    u16* bd = &Bs[buf][lrow * LSTR + lhalf];
    const u16* bg = Wt + ((size_t)((p << 10) + n0 + lrow)) * 1024 + kc + lhalf;
#ifdef HAVE_ASYNC_LDS
    if (((unsigned)yy < 64u) && ((unsigned)xx < 64u)) {
      const u16* ag = Xt + ((size_t)(yy * 64 + xx)) * 1024 + kc + lhalf;
      ASYNC_CP16(ad, ag);
      ASYNC_CP16(ad + 8, ag + 8);
    } else {
      uint4 z = {0, 0, 0, 0};
      *(uint4*)ad = z;
      *(uint4*)(ad + 8) = z;
    }
    ASYNC_CP16(bd, bg);
    ASYNC_CP16(bd + 8, bg + 8);
#else
    uint4 a0 = {0, 0, 0, 0}, a1 = {0, 0, 0, 0};
    if (((unsigned)yy < 64u) && ((unsigned)xx < 64u)) {
      const uint4* gp = (const uint4*)(Xt + ((size_t)(yy * 64 + xx)) * 1024 + kc + lhalf);
      a0 = gp[0]; a1 = gp[1];
    }
    const uint4* gq = (const uint4*)bg;
    uint4 b0 = gq[0], b1 = gq[1];
    *(uint4*)ad = a0; *(uint4*)(ad + 8) = a1;
    *(uint4*)bd = b0; *(uint4*)(bd + 8) = b1;
#endif
  };

  auto compute = [&](int buf) {
    F a[4], b[2];
#pragma unroll
    for (int i = 0; i < 4; ++i) {
      const int r = (wm + i * 16 + am) * LSTR;
      a[i].u[0] = *(const uint4*)&As[buf][r + 2 * ak];
      a[i].u[1] = *(const uint4*)&As[buf][r + 2 * ak + 8];
    }
#pragma unroll
    for (int j = 0; j < 2; ++j) {
      const int r = (wn + j * 16 + am) * LSTR;
      b[j].u[0] = *(const uint4*)&Bs[buf][r + bk];
      b[j].u[1] = *(const uint4*)&Bs[buf][r + bk + 8];
    }
#pragma unroll
    for (int i = 0; i < 4; ++i)
#pragma unroll
      for (int j = 0; j < 2; ++j)
        acc[i][j] = __builtin_amdgcn_wmma_f32_16x16x32_bf16(
            false, a[i].v, false, b[j].v, (short)0, acc[i][j], false, false);
  };

  const int NSTEP = 9 * 32;                          // 288 K-steps of 32
  stage(0, 0);
  for (int s = 0; s < NSTEP; ++s) {
    WAIT_ASYNC();                                    // buf[s&1] fully landed (this wave)
    __syncthreads();                                 // all waves landed + prior reads done
    if (s + 1 < NSTEP) stage(s + 1, (s + 1) & 1);    // prefetch next step async
    compute(s & 1);
  }

  const int mhi = (lane >> 4) * 8;                   // C/D layout: VGPR r -> M=r / r+8
#pragma unroll
  for (int i = 0; i < 4; ++i) {
#pragma unroll
    for (int j = 0; j < 2; ++j) {
      const int n = n0 + wn + j * 16 + am;
      const float bv = bias[n];
#pragma unroll
      for (int r = 0; r < 8; ++r) {
        const int m = m0 + wm + i * 16 + r + mhi;
        float v = acc[i][j][r] + bv;
        v = v > 0.f ? v : 0.f;                       // ReLU
        Ht[(size_t)m * 1024 + n] = f2bf(v);
      }
    }
  }
}

// ---------------- 1x1 heads GEMM: raw[4096][80] = Ht(4096x1024) * Wsl^T ----------------
__global__ __launch_bounds__(256) void k_head(
    const u16* __restrict__ Ht, const u16* __restrict__ Wsl,
    const float* __restrict__ sb, const float* __restrict__ lb,
    float* __restrict__ raw) {
  __shared__ u16 As[2][128 * LSTR];
  __shared__ u16 Bs[2][80 * LSTR];
  const int tid = threadIdx.x, lane = tid & 31, wave = tid >> 5;
  const int m0 = blockIdx.x * 128;
  v8f zero = {0.f, 0.f, 0.f, 0.f, 0.f, 0.f, 0.f, 0.f};
  v8f acc[5];
#pragma unroll
  for (int j = 0; j < 5; ++j) acc[j] = zero;
  const int lrow = tid >> 1, lhalf = (tid & 1) * 16;
  const int am = lane & 15, ak = (lane >> 4) * 8, bk = (lane >> 4) * 16;
  union F { uint4 u[2]; v16bf v; };

  auto stage = [&](int s, int buf) {
    const int kc = s << 5;
    u16* ad = &As[buf][lrow * LSTR + lhalf];
    const u16* ag = Ht + (size_t)(m0 + lrow) * 1024 + kc + lhalf;
#ifdef HAVE_ASYNC_LDS
    ASYNC_CP16(ad, ag);
    ASYNC_CP16(ad + 8, ag + 8);
    if (lrow < 80) {
      u16* bd = &Bs[buf][lrow * LSTR + lhalf];
      const u16* bg = Wsl + (size_t)lrow * 1024 + kc + lhalf;
      ASYNC_CP16(bd, bg);
      ASYNC_CP16(bd + 8, bg + 8);
    }
#else
    const uint4* gp = (const uint4*)ag;
    uint4 a0 = gp[0], a1 = gp[1];
    *(uint4*)ad = a0; *(uint4*)(ad + 8) = a1;
    if (lrow < 80) {
      const uint4* gq = (const uint4*)(Wsl + (size_t)lrow * 1024 + kc + lhalf);
      uint4 b0 = gq[0], b1 = gq[1];
      u16* bd = &Bs[buf][lrow * LSTR + lhalf];
      *(uint4*)bd = b0; *(uint4*)(bd + 8) = b1;
    }
#endif
  };

  auto compute = [&](int buf) {
    F a;
    const int r = (wave * 16 + am) * LSTR;
    a.u[0] = *(const uint4*)&As[buf][r + 2 * ak];
    a.u[1] = *(const uint4*)&As[buf][r + 2 * ak + 8];
#pragma unroll
    for (int j = 0; j < 5; ++j) {
      F b;
      const int rb = (j * 16 + am) * LSTR;
      b.u[0] = *(const uint4*)&Bs[buf][rb + bk];
      b.u[1] = *(const uint4*)&Bs[buf][rb + bk + 8];
      acc[j] = __builtin_amdgcn_wmma_f32_16x16x32_bf16(
          false, a.v, false, b.v, (short)0, acc[j], false, false);
    }
  };

  stage(0, 0);
  for (int s = 0; s < 32; ++s) {
    WAIT_ASYNC();
    __syncthreads();
    if (s + 1 < 32) stage(s + 1, (s + 1) & 1);
    compute(s & 1);
  }

  const int mhi = (lane >> 4) * 8;
#pragma unroll
  for (int j = 0; j < 5; ++j) {
    const int n = j * 16 + am;
    const float bv = (n < 15) ? sb[n] : (n < 75 ? lb[n - 15] : 0.f);
#pragma unroll
    for (int r = 0; r < 8; ++r) {
      const int m = m0 + wave * 16 + r + mhi;
      raw[(size_t)m * 80 + n] = acc[j][r] + bv;
    }
  }
}

// ---------------- anchor decode + clip + sigmoid + min-size ----------------
__global__ void k_decode(const float* __restrict__ raw, float* __restrict__ boxes,
                         float* __restrict__ keys, int* __restrict__ vals) {
  int i = blockIdx.x * 256 + threadIdx.x;            // 65536
  if (i >= 61440) { keys[i] = -INFINITY; vals[i] = i; return; }
  int a = i % 15, pos = i / 15;
  int y = pos >> 6, x = pos & 63;
  int r = a / 5, s = a % 5;
  float ratio = 0.5f * (float)(1 << r);              // 0.5,1,2
  float scale = (float)(2 << s);                     // 2,4,8,16,32
  float wsr = roundf(sqrtf(256.0f / ratio));
  float hsr = roundf(wsr * ratio);
  float w2 = wsr * scale, h2 = hsr * scale;
  float cx = 7.5f + 16.0f * (float)x, cy = 7.5f + 16.0f * (float)y;
  float ax1 = cx - 0.5f * (w2 - 1.f), ax2 = cx + 0.5f * (w2 - 1.f);
  float ay1 = cy - 0.5f * (h2 - 1.f), ay2 = cy + 0.5f * (h2 - 1.f);
  float aw = ax2 - ax1, ah = ay2 - ay1;
  float axc = ax1 + 0.5f * aw, ayc = ay1 + 0.5f * ah;
  const float* rp = raw + (size_t)pos * 80;
  float sc = 1.f / (1.f + expf(-rp[a]));
  float dx = rp[15 + a * 4 + 0], dy = rp[15 + a * 4 + 1];
  float dw = rp[15 + a * 4 + 2], dh = rp[15 + a * 4 + 3];
  float ox = dx * aw + axc, oy = dy * ah + ayc;
  float ow = expf(fminf(dw, 4.14f)) * aw * 0.5f;
  float oh = expf(fminf(dh, 4.14f)) * ah * 0.5f;
  float x1 = fminf(fmaxf(ox - ow, 0.f), 1023.f);
  float y1 = fminf(fmaxf(oy - oh, 0.f), 1023.f);
  float x2 = fminf(fmaxf(ox + ow, 0.f), 1023.f);
  float y2 = fminf(fmaxf(oy + oh, 0.f), 1023.f);
  if ((x2 - x1) < 16.f || (y2 - y1) < 16.f) { sc = -1.f; x1 = y1 = x2 = y2 = -1.f; }
  boxes[i * 4 + 0] = x1; boxes[i * 4 + 1] = y1;
  boxes[i * 4 + 2] = x2; boxes[i * 4 + 3] = y2;
  keys[i] = sc; vals[i] = i;
}

// ---------------- global bitonic sort step (descending) ----------------
__global__ void k_bitonic(float* keys, int* vals, int j, int k) {
  int i = blockIdx.x * 256 + threadIdx.x;
  int ixj = i ^ j;
  if (ixj <= i) return;
  float ki = keys[i], kj = keys[ixj];
  bool descSeg = ((i & k) == 0);
  bool doswap = descSeg ? (ki < kj) : (ki > kj);
  if (doswap) {
    keys[i] = kj; keys[ixj] = ki;
    int vi = vals[i]; vals[i] = vals[ixj]; vals[ixj] = vi;
  }
}

__global__ void k_gather(const float* __restrict__ keys, const int* __restrict__ vals,
                         const float* __restrict__ boxes, float* __restrict__ topS,
                         float* __restrict__ topB, float* __restrict__ areas) {
  int i = blockIdx.x * 256 + threadIdx.x;
  if (i >= 6000) return;
  int o = vals[i];
  float x1 = -1.f, y1 = -1.f, x2 = -1.f, y2 = -1.f, sc = -1.f;
  if (o < 61440) {
    x1 = boxes[o * 4 + 0]; y1 = boxes[o * 4 + 1];
    x2 = boxes[o * 4 + 2]; y2 = boxes[o * 4 + 3];
    sc = keys[i];
  }
  topS[i] = sc;
  topB[i * 4 + 0] = x1; topB[i * 4 + 1] = y1;
  topB[i * 4 + 2] = x2; topB[i * 4 + 3] = y2;
  areas[i] = (x2 - x1) * (y2 - y1);
}

// ---------------- NMS suppression bitmask ----------------
__global__ void k_nmsmask(const float* __restrict__ topB, const float* __restrict__ areas,
                          u64* __restrict__ mask) {
  __shared__ float cb[64 * 4];
  __shared__ float ca[64];
  const int t  = threadIdx.x;
  const int c0 = blockIdx.x * 64;
  const int r  = blockIdx.y * 64 + t;
  if (c0 + t < 6000) {
    cb[t * 4 + 0] = topB[(c0 + t) * 4 + 0];
    cb[t * 4 + 1] = topB[(c0 + t) * 4 + 1];
    cb[t * 4 + 2] = topB[(c0 + t) * 4 + 2];
    cb[t * 4 + 3] = topB[(c0 + t) * 4 + 3];
    ca[t] = areas[c0 + t];
  }
  __syncthreads();
  if (r >= 6000) return;
  float rx1 = topB[r * 4 + 0], ry1 = topB[r * 4 + 1];
  float rx2 = topB[r * 4 + 2], ry2 = topB[r * 4 + 3];
  float ra = areas[r];
  u64 bits = 0;
  int lim = 6000 - c0; if (lim > 64) lim = 64;
  for (int jj = 0; jj < lim; ++jj) {
    int c = c0 + jj;
    if (c == r) continue;
    float iw = fmaxf(fminf(rx2, cb[jj * 4 + 2]) - fmaxf(rx1, cb[jj * 4 + 0]), 0.f);
    float ih = fmaxf(fminf(ry2, cb[jj * 4 + 3]) - fmaxf(ry1, cb[jj * 4 + 1]), 0.f);
    float inter = iw * ih;
    float iou = inter / (ra + ca[jj] - inter + 1e-12f);
    if (iou > 0.7f) bits |= (1ull << jj);
  }
  mask[(size_t)r * NB + blockIdx.x] = bits;
}

// ---------------- sequential greedy scan + emit top-300 ----------------
__global__ void k_finalize(const u64* __restrict__ mask, const float* __restrict__ topS,
                           const float* __restrict__ topB, float* __restrict__ out) {
  __shared__ u64 supp[NB];
  __shared__ int keepFlag;
  __shared__ char kept[6000];
  const int tid = threadIdx.x;
  if (tid < NB) supp[tid] = 0;
  for (int i = 0; i < 6000; ++i) {
    __syncthreads();
    if (tid == 0) keepFlag = ((supp[i >> 6] >> (i & 63)) & 1ull) ? 0 : 1;
    __syncthreads();
    if (keepFlag && tid < NB) supp[tid] |= mask[(size_t)i * NB + tid];
    if (tid == 0) kept[i] = (char)keepFlag;
  }
  __syncthreads();
  if (tid == 0) {
    int cnt = 0;
    for (int i = 0; i < 6000 && cnt < 300; ++i) {
      if (kept[i]) {
        out[cnt * 5 + 0] = topS[i];
        out[cnt * 5 + 1] = topB[i * 4 + 0];
        out[cnt * 5 + 2] = topB[i * 4 + 1];
        out[cnt * 5 + 3] = topB[i * 4 + 2];
        out[cnt * 5 + 4] = topB[i * 4 + 3];
        cnt++;
      }
    }
    for (; cnt < 300; ++cnt) {
      out[cnt * 5 + 0] = -1.f; out[cnt * 5 + 1] = -1.f; out[cnt * 5 + 2] = -1.f;
      out[cnt * 5 + 3] = -1.f; out[cnt * 5 + 4] = -1.f;
    }
  }
}

extern "C" void kernel_launch(void* const* d_in, const int* in_sizes, int n_in,
                              void* d_out, int out_size, void* d_ws, size_t ws_size,
                              hipStream_t stream) {
  (void)in_sizes; (void)n_in; (void)out_size; (void)ws_size;
  const float* x       = (const float*)d_in[1];
  const float* conv_w  = (const float*)d_in[2];
  const float* conv_b  = (const float*)d_in[3];
  const float* score_w = (const float*)d_in[4];
  const float* score_b = (const float*)d_in[5];
  const float* loc_w   = (const float*)d_in[6];
  const float* loc_b   = (const float*)d_in[7];

  char* ws = (char*)d_ws;
  size_t off = 0;
  auto take = [&](size_t bytes) -> char* {
    char* p = ws + off;
    off = (off + bytes + 255) & ~(size_t)255;
    return p;
  };
  u16*   Wt    = (u16*)  take(9ull * 1024 * 1024 * 2);   // 18.9 MB
  u16*   Xt    = (u16*)  take(4096ull * 1024 * 2);       //  8.4 MB
  u16*   Ht    = (u16*)  take(4096ull * 1024 * 2);       //  8.4 MB
  u16*   Wsl   = (u16*)  take(80ull * 1024 * 2);
  float* raw   = (float*)take(4096ull * 80 * 4);
  float* boxes = (float*)take(61440ull * 4 * 4);
  float* keys  = (float*)take(65536ull * 4);
  int*   vals  = (int*)  take(65536ull * 4);
  float* topS  = (float*)take(6000ull * 4);
  float* topB  = (float*)take(6000ull * 16);
  float* areas = (float*)take(6000ull * 4);
  u64*   mask  = (u64*)  take(6000ull * NB * 8);         //  4.5 MB

  k_prep_w  <<<dim3(36864), dim3(256), 0, stream>>>(conv_w, Wt);
  k_prep_x  <<<dim3(16384), dim3(256), 0, stream>>>(x, Xt);
  k_prep_wsl<<<dim3(320),   dim3(256), 0, stream>>>(score_w, loc_w, Wsl);

  k_conv3x3 <<<dim3(8, 32), dim3(256), 0, stream>>>(Xt, Wt, conv_b, Ht);
  k_head    <<<dim3(32),    dim3(256), 0, stream>>>(Ht, Wsl, score_b, loc_b, raw);
  k_decode  <<<dim3(256),   dim3(256), 0, stream>>>(raw, boxes, keys, vals);

  for (int k = 2; k <= 65536; k <<= 1)
    for (int j = k >> 1; j > 0; j >>= 1)
      k_bitonic<<<dim3(256), dim3(256), 0, stream>>>(keys, vals, j, k);

  k_gather  <<<dim3(24),      dim3(256), 0, stream>>>(keys, vals, boxes, topS, topB, areas);
  k_nmsmask <<<dim3(NB, NB),  dim3(64),  0, stream>>>(topB, areas, mask);
  k_finalize<<<dim3(1),       dim3(128), 0, stream>>>(mask, topS, topB, (float*)d_out);
}
